// GCN_76922864271731
// MI455X (gfx1250) — compile-verified, matrix operand order checked by reading, not run on hardware
//
#include <hip/hip_runtime.h>
#include <hip/hip_bf16.h>

typedef float    v2f  __attribute__((ext_vector_type(2)));
typedef float    v8f  __attribute__((ext_vector_type(8)));
typedef _Float16 v16h __attribute__((ext_vector_type(16)));

#define NN   1024
#define HH   64
#define SS   2
#define DD   320          // (GDEP*S+1)*h
#define BETA 0.05f

// ---------------------------------------------------------------------------
// Kernel 1: Anorm[s,i,j] = (A[s,i,j] + (i==j)) / row_sum    (2048 rows)
// One float4 per thread (1024 elem / 256 thr); row cached in registers so A
// is read exactly once (8 MB instead of 16 MB), all traffic b128.
// ---------------------------------------------------------------------------
__global__ void gcn_norm_rows(const float4* __restrict__ A4, float4* __restrict__ An4) {
    const int row = blockIdx.x;            // s*N + i
    const int i   = row & (NN - 1);
    const int tid = threadIdx.x;

    float4 v = A4[(size_t)row * (NN / 4) + tid];
    if (tid == (i >> 2)) ((float*)&v)[i & 3] += 1.f;

    float part = v.x + v.y + v.z + v.w;
    for (int off = 16; off > 0; off >>= 1)
        part += __shfl_down(part, off, 32);

    __shared__ float red[8];
    __shared__ float s_inv;
    if ((tid & 31) == 0) red[tid >> 5] = part;
    __syncthreads();
    if (tid == 0) {
        float t = 0.f;
        #pragma unroll
        for (int k = 0; k < 8; ++k) t += red[k];
        s_inv = 1.f / t;
    }
    __syncthreads();
    const float inv = s_inv;
    v.x *= inv; v.y *= inv; v.z *= inv; v.w *= inv;
    An4[(size_t)row * (NN / 4) + tid] = v;
}

// ---------------------------------------------------------------------------
// Kernel 2: X[i][0:64] = H_in[i][:]   (b128)
// ---------------------------------------------------------------------------
__global__ void gcn_copy_hin(const float4* __restrict__ Hin4, float4* __restrict__ X4) {
    const int i = blockIdx.x;
    X4[(size_t)i * (DD / 4) + threadIdx.x] = Hin4[(size_t)i * (HH / 4) + threadIdx.x];
}

// ---------------------------------------------------------------------------
// Kernel 3 (launched twice): one propagation step.
//   diff[s,i,h] = sum_j Anorm[s,i,j] * Hprev[s,j,h] * C[i,j,h]
//   Hnew = BETA*Hprev + (1-BETA)*diff
// One block per row i: 256 thr = 16 h-quads (float4 along h) x 16 j-slices.
// All C / H / X traffic is global_load_b128; C[i,:,:] (256 KB) read once for
// both supports. Offsets prev4/out4 are float4 column offsets into X rows.
// ---------------------------------------------------------------------------
__global__ void gcn_prop(const float* __restrict__ An, const float4* __restrict__ C4,
                         const float4* __restrict__ Hin4, float4* __restrict__ X4,
                         int prev4, int out4, int first) {
    const int i  = blockIdx.x;
    const int hq = threadIdx.x & 15;     // h-quad: channels 4*hq .. 4*hq+3
    const int jt = threadIdx.x >> 4;     // 0..15

    const float4* __restrict__ Crow = C4 + (size_t)i * NN * (HH / 4);
    const float*  __restrict__ A0   = An + (size_t)i * NN;                   // s=0
    const float*  __restrict__ A1   = An + (size_t)NN * NN + (size_t)i * NN; // s=1

    float4 acc0 = make_float4(0.f, 0.f, 0.f, 0.f);
    float4 acc1 = make_float4(0.f, 0.f, 0.f, 0.f);

    for (int j = jt; j < NN; j += 16) {
        if (j + 64 < NN)   // stream-ahead hint -> global_prefetch_b8
            __builtin_prefetch(&Crow[(size_t)(j + 64) * (HH / 4) + hq], 0, 0);
        const float4 c = Crow[(size_t)j * (HH / 4) + hq];
        float4 h0, h1;
        if (first) {
            h0 = Hin4[(size_t)j * (HH / 4) + hq];
            h1 = h0;
        } else {
            h0 = X4[(size_t)j * (DD / 4) + prev4 + hq];        // L2-resident
            h1 = X4[(size_t)j * (DD / 4) + prev4 + 16 + hq];
        }
        const float a0 = A0[j];
        const float a1 = A1[j];
        acc0.x = fmaf(a0 * h0.x, c.x, acc0.x);
        acc0.y = fmaf(a0 * h0.y, c.y, acc0.y);
        acc0.z = fmaf(a0 * h0.z, c.z, acc0.z);
        acc0.w = fmaf(a0 * h0.w, c.w, acc0.w);
        acc1.x = fmaf(a1 * h1.x, c.x, acc1.x);
        acc1.y = fmaf(a1 * h1.y, c.y, acc1.y);
        acc1.z = fmaf(a1 * h1.z, c.z, acc1.z);
        acc1.w = fmaf(a1 * h1.w, c.w, acc1.w);
    }

    __shared__ float4 red[2][16][16];    // [s][jt][hq], 8 KB
    red[0][jt][hq] = acc0;
    red[1][jt][hq] = acc1;
    __syncthreads();

    if (threadIdx.x < 32) {
        const int s   = threadIdx.x >> 4;
        const int hq2 = threadIdx.x & 15;
        float4 d = make_float4(0.f, 0.f, 0.f, 0.f);
        #pragma unroll
        for (int t = 0; t < 16; ++t) {
            const float4 r = red[s][t][hq2];
            d.x += r.x; d.y += r.y; d.z += r.z; d.w += r.w;
        }
        float4 hold;
        if (first) hold = Hin4[(size_t)i * (HH / 4) + hq2];
        else       hold = X4[(size_t)i * (DD / 4) + prev4 + s * 16 + hq2];
        float4 hn;
        hn.x = BETA * hold.x + (1.f - BETA) * d.x;
        hn.y = BETA * hold.y + (1.f - BETA) * d.y;
        hn.z = BETA * hold.z + (1.f - BETA) * d.z;
        hn.w = BETA * hold.w + (1.f - BETA) * d.w;
        X4[(size_t)i * (DD / 4) + out4 + s * 16 + hq2] = hn;
    }
}

// ---------------------------------------------------------------------------
// Kernel 4: Out[1024,64] = X[1024,320] @ W.T + b  via V_WMMA_F32_16X16X4_F32.
// One wave per 16x16 output tile; K loop of 320 in steps of 4.
// A layout (16x4 f32):  lane L -> m = L%16;  element v -> K = kbase + 2*(L/16)+v
// B layout (4x16 f32):  lane L -> n = L%16;  element v -> K = kbase + 2*(L/16)+v
// D layout (16x16 f32): element v at lane L -> (m = v + 8*(L/16), n = L%16)
// ---------------------------------------------------------------------------
__global__ void gcn_proj(const float* __restrict__ X, const float* __restrict__ W,
                         const float* __restrict__ b, float* __restrict__ Out) {
    const int lane = threadIdx.x;        // 0..31 (wave32)
    const int m0   = blockIdx.x * 16;    // output row tile
    const int n0   = blockIdx.y * 16;    // output col tile
    const int grp  = lane >> 4;          // 0/1
    const int lm   = lane & 15;

    v8f acc;
    const float bias = b[n0 + lm];       // D col for this lane is n0+lm in every slot
    #pragma unroll
    for (int v = 0; v < 8; ++v) acc[v] = bias;

#if __has_builtin(__builtin_amdgcn_wmma_f32_16x16x4_f32)
    for (int k = 0; k < DD; k += 4) {
        const int ka = k + 2 * grp;
        const float* xp = X + (size_t)(m0 + lm) * DD + ka;   // A: X rows
        const float* wp = W + (size_t)(n0 + lm) * DD + ka;   // B[k][n] = W[n][k]
        v2f av, bv;
        av[0] = xp[0]; av[1] = xp[1];
        bv[0] = wp[0]; bv[1] = wp[1];
        acc = __builtin_amdgcn_wmma_f32_16x16x4_f32(
            /*neg_a=*/false, av, /*neg_b=*/false, bv,
            /*c_mod=*/(short)0, acc, /*reuse_a=*/false, /*reuse_b=*/false);
    }
#else
    // Fallback: codegen-confirmed f16 WMMA (16x16x32), on-the-fly f32->f16.
    for (int k = 0; k < DD; k += 32) {
        v16h av, bv;
        #pragma unroll
        for (int e = 0; e < 16; ++e) {
            const int v  = e >> 1, hl = e & 1;
            const int kk = ((v < 4) ? 2 * v : 16 + 2 * (v - 4)) + 8 * grp + hl;
            av[e] = (_Float16)X[(size_t)(m0 + lm) * DD + k + kk];
            bv[e] = (_Float16)W[(size_t)(n0 + lm) * DD + k + kk];
        }
        acc = __builtin_amdgcn_wmma_f32_16x16x32_f16(
            false, av, false, bv, (short)0, acc, false, false);
    }
#endif

    #pragma unroll
    for (int v = 0; v < 8; ++v) {
        const int m = v + 8 * grp;
        Out[(size_t)(m0 + m) * HH + n0 + lm] = acc[v];
    }
}

// ---------------------------------------------------------------------------
extern "C" void kernel_launch(void* const* d_in, const int* in_sizes, int n_in,
                              void* d_out, int out_size, void* d_ws, size_t ws_size,
                              hipStream_t stream) {
    (void)in_sizes; (void)n_in; (void)out_size; (void)ws_size;

    const float* H_in = (const float*)d_in[0];   // [1024, 64]
    const float* A    = (const float*)d_in[1];   // [2, 1024, 1024]
    const float* C    = (const float*)d_in[2];   // [1024, 1024, 64]
    const float* W    = (const float*)d_in[3];   // [64, 320]
    const float* b    = (const float*)d_in[4];   // [64]
    float*       Out  = (float*)d_out;           // [1024, 64]

    float* ws    = (float*)d_ws;
    float* Anorm = ws;                            // 2*1024*1024 f32 (8 MB)
    float* X     = ws + (size_t)SS * NN * NN;     // 1024*320    f32 (1.3 MB)

    gcn_norm_rows<<<SS * NN, 256, 0, stream>>>((const float4*)A, (float4*)Anorm);
    gcn_copy_hin <<<NN, HH / 4, 0, stream>>>((const float4*)H_in, (float4*)X);
    // step 1: Hprev = broadcast(H_in) -> H1 in X cols [64,192)  (float4 off 16)
    gcn_prop<<<NN, 256, 0, stream>>>(Anorm, (const float4*)C, (const float4*)H_in,
                                     (float4*)X, /*prev4=*/0,  /*out4=*/16, /*first=*/1);
    // step 2: Hprev = H1 -> H2 in X cols [192,320)              (float4 off 48)
    gcn_prop<<<NN, 256, 0, stream>>>(Anorm, (const float4*)C, (const float4*)H_in,
                                     (float4*)X, /*prev4=*/16, /*out4=*/48, /*first=*/0);
    // projection (WMMA)
    gcn_proj<<<dim3(NN / 16, HH / 16), 32, 0, stream>>>(X, W, b, Out);
}
